// BlockwiseQuantLinear_55259049230824
// MI455X (gfx1250) — 8-bit matrix operand order repaired and checked by reading, compiled, not run on hardware
//
#include <hip/hip_runtime.h>
#include <hip/hip_bf16.h>
#include <stdint.h>

#define QBLOCK   128
#define FP8_MAX  448.0f
#define M_DIM    65536
#define K_DIM    1024
#define N_DIM    1024
#define KB_CNT   (K_DIM / QBLOCK)
#define NB_CNT   (N_DIM / QBLOCK)

typedef __attribute__((ext_vector_type(16))) int   v16i;
typedef __attribute__((ext_vector_type(8)))  float v8f;
typedef __attribute__((ext_vector_type(4)))  float v4f;

__global__ __launch_bounds__(256) void quant_x_kernel(const float* __restrict__ x,
                                                      uint8_t* __restrict__ xq,
                                                      float* __restrict__ xs)
{
    const int lane = threadIdx.x & 31;
    const int wave = threadIdx.x >> 5;
    const size_t task = (size_t)blockIdx.x * 8 + wave;

    const v4f* src = (const v4f*)(x + task * QBLOCK) + lane;
    v4f v = __builtin_nontemporal_load(src);

    float am = fmaxf(fmaxf(fabsf(v.x), fabsf(v.y)), fmaxf(fabsf(v.z), fabsf(v.w)));
    #pragma unroll
    for (int off = 16; off > 0; off >>= 1)
        am = fmaxf(am, __shfl_xor(am, off, 32));

    const float scale = fmaxf(am / FP8_MAX, 1e-12f);
    const float inv   = 1.0f / scale;

    int packed = __builtin_amdgcn_cvt_pk_fp8_f32(v.x * inv, v.y * inv, 0, false);
    packed     = __builtin_amdgcn_cvt_pk_fp8_f32(v.z * inv, v.w * inv, packed, true);

    *(int*)(xq + task * QBLOCK + lane * 4) = packed;
    if (lane == 0)
        xs[task] = scale;
}

__global__ __launch_bounds__(256) void fp8_gemm_kernel(const uint8_t* __restrict__ xq,
                                                       const float*   __restrict__ xs,
                                                       const uint8_t* __restrict__ wq,
                                                       const float*   __restrict__ wscale,
                                                       float*         __restrict__ out)
{
    const int lane   = threadIdx.x & 31;
    const int wave   = blockIdx.x * 8 + (threadIdx.x >> 5);
    const int n_tile = wave & (N_DIM / 32 - 1);
    const int m_tile = wave / (N_DIM / 32);
    const int m_base = m_tile * 32;
    const int n_base = n_tile * 32;
    const int nb     = n_base >> 7;
    const int lrow   = lane & 15;
    const int hi     = lane >> 4;

    const uint8_t* pa0 = wq + (size_t)(n_base + lrow) * K_DIM + hi * 16;
    const uint8_t* pa1 = pa0 + (size_t)16 * K_DIM;
    const uint8_t* pb0 = xq + (size_t)(m_base + lrow) * K_DIM + hi * 16;
    const uint8_t* pb1 = pb0 + (size_t)16 * K_DIM;

    const float* pxs0 = xs + (size_t)(m_base + lrow) * KB_CNT;
    const float* pxs1 = pxs0 + (size_t)16 * KB_CNT;

    v8f acc00 = {}, acc01 = {}, acc10 = {}, acc11 = {};

    #pragma unroll
    for (int kb = 0; kb < KB_CNT; ++kb) {
        const int ko = kb * QBLOCK;

        v16i a0, a1, b0, b1;
        #pragma unroll
        for (int c = 0; c < 4; ++c) {
            int4 d0 = *(const int4*)(pa0 + ko + c * 32);
            int4 d1 = *(const int4*)(pa1 + ko + c * 32);
            a0[4 * c + 0] = d0.x; a0[4 * c + 1] = d0.y;
            a0[4 * c + 2] = d0.z; a0[4 * c + 3] = d0.w;
            a1[4 * c + 0] = d1.x; a1[4 * c + 1] = d1.y;
            a1[4 * c + 2] = d1.z; a1[4 * c + 3] = d1.w;
        }
        #pragma unroll
        for (int c = 0; c < 4; ++c) {
            int4 e0 = *(const int4*)(pb0 + ko + c * 32);
            int4 e1 = *(const int4*)(pb1 + ko + c * 32);
            b0[4 * c + 0] = e0.x; b0[4 * c + 1] = e0.y;
            b0[4 * c + 2] = e0.z; b0[4 * c + 3] = e0.w;
            b1[4 * c + 0] = e1.x; b1[4 * c + 1] = e1.y;
            b1[4 * c + 2] = e1.z; b1[4 * c + 3] = e1.w;
        }

        const float ws = wscale[nb * KB_CNT + kb];
        const float s0 = pxs0[kb] * ws;
        const float s1 = pxs1[kb] * ws;

        v8f z = {};
        v8f p00 = __builtin_amdgcn_wmma_f32_16x16x128_fp8_fp8(a0, b0, (short)0, z, false, false);
        v8f p01 = __builtin_amdgcn_wmma_f32_16x16x128_fp8_fp8(a0, b1, (short)0, z, false, false);
        v8f p10 = __builtin_amdgcn_wmma_f32_16x16x128_fp8_fp8(a1, b0, (short)0, z, false, false);
        v8f p11 = __builtin_amdgcn_wmma_f32_16x16x128_fp8_fp8(a1, b1, (short)0, z, false, false);

        #pragma unroll
        for (int v = 0; v < 8; ++v) {
            acc00[v] += s0 * p00[v];
            acc01[v] += s1 * p01[v];
            acc10[v] += s0 * p10[v];
            acc11[v] += s1 * p11[v];
        }
    }

    const v8f* accs[2][2] = { { &acc00, &acc01 }, { &acc10, &acc11 } };
    #pragma unroll
    for (int ai = 0; ai < 2; ++ai) {
        #pragma unroll
        for (int bi = 0; bi < 2; ++bi) {
            const v8f a = *accs[ai][bi];
            float* dst = out + (size_t)(m_base + bi * 16 + lrow) * N_DIM
                             + (n_base + ai * 16 + hi * 8);
            v4f lo = { a[0], a[1], a[2], a[3] };
            v4f hh = { a[4], a[5], a[6], a[7] };
            __builtin_nontemporal_store(lo, (v4f*)dst);
            __builtin_nontemporal_store(hh, (v4f*)(dst + 4));
        }
    }
}

extern "C" void kernel_launch(void* const* d_in, const int* in_sizes, int n_in,
                              void* d_out, int out_size, void* d_ws, size_t ws_size,
                              hipStream_t stream)
{
    const float*   x       = (const float*)d_in[0];
    const uint8_t* weight  = (const uint8_t*)d_in[1];
    const float*   w_scale = (const float*)d_in[2];
    float*         out     = (float*)d_out;

    uint8_t* xq = (uint8_t*)d_ws;
    float*   xs = (float*)((uint8_t*)d_ws + (size_t)M_DIM * K_DIM);

    quant_x_kernel<<<(M_DIM * KB_CNT) / 8, 256, 0, stream>>>(x, xq, xs);

    fp8_gemm_kernel<<<((M_DIM / 32) * (N_DIM / 32)) / 8, 256, 0, stream>>>(
        xq, xs, weight, w_scale, out);
}
